// YatConvBlock_47210280518315
// MI455X (gfx1250) — matrix operand order repaired and checked by reading, not compiled, not run on hardware
//
#include <hip/hip_runtime.h>

typedef __attribute__((ext_vector_type(16))) __bf16 v16bf;
typedef __attribute__((ext_vector_type(8)))  __bf16 v8bf;
typedef __attribute__((ext_vector_type(8)))  float  v8f;
typedef __attribute__((ext_vector_type(4)))  unsigned int u32x4;
typedef __attribute__((ext_vector_type(8)))  int i32x8;
typedef __attribute__((ext_vector_type(4)))  int i32x4;

#define WMMA_BF16(A, B, C) \
    __builtin_amdgcn_wmma_f32_16x16x32_bf16(false, (A), false, (B), (short)0, (C), false, false)

// A fragment (16x32 bf16): lane (half = lane>>4) holds K = half*8+j (j=0..7) and K = 16+half*8+j.
// With channels contiguous in LDS this is two 16-byte loads.
__device__ __forceinline__ v16bf make_afrag(const __bf16* p) {
    v8bf lo = *(const v8bf*)(p);
    v8bf hi = *(const v8bf*)(p + 16);
    v16bf a;
#pragma unroll
    for (int j = 0; j < 8; ++j) { a[j] = lo[j]; a[j + 8] = hi[j]; }
    return a;
}

// TDM: load one row of `elems` bf16 from global into LDS at byte offset lds_off.
// pad_enable with interval=128 DWORDs (512B) and amount=4 DWORDs (16B) reproduces the
// 264-element LDS stride per 256-channel chunk (bank-conflict padding) in hardware.
__device__ __forceinline__ void tdm_load_row(unsigned lds_off, const void* gsrc, unsigned elems) {
    const unsigned long long ga = (unsigned long long)gsrc;
    u32x4 g0 = {};
    g0[0] = 1u;                                                  // count=1, user descriptor
    g0[1] = lds_off;                                             // lds_addr [63:32]
    g0[2] = (unsigned)(ga & 0xffffffffu);                        // global_addr [95:64]
    g0[3] = (unsigned)((ga >> 32) & 0x01ffffffu) | (2u << 30);   // addr[56:32] | type=2
    i32x8 g1 = {};
    g1[0] = (int)((1u << 16) |                                   // data_size = 2 bytes
                  (1u << 20) |                                   // pad_enable
                  (6u << 22) |                                   // pad_interval: 128 DWORDs
                  (3u << 25));                                   // pad_amount: 4 DWORDs
    g1[1] = (int)((elems & 0xffffu) << 16);                      // tensor_dim0[15:0]
    g1[2] = (int)(((elems >> 16) & 0xffffu) | (1u << 16));       // tensor_dim0 hi | tensor_dim1=1
    g1[3] = (int)((elems & 0xffffu) << 16);                      // tile_dim0
    g1[4] = 1;                                                   // tile_dim1=1, tile_dim2=0
    g1[5] = (int)elems;                                          // tensor_dim0_stride (unused, 1 row)
    const i32x4 z4 = {};
    const i32x8 z8 = {};
    __builtin_amdgcn_tensor_load_to_lds(g0, g1, z4, z4, z8, 0);
}

// ---------------- kernel 0a: per-cout sum of squares of w_yat ----------------
__global__ __launch_bounds__(256) void k_wsq(const float* __restrict__ w,
                                             float* __restrict__ wsq) {
    const int t = threadIdx.x;  // cout
    float s = 0.f;
    for (int k = 0; k < 1152; ++k) {
        float v = w[(size_t)k * 256 + t];
        s += v * v;
    }
    wsq[t] = s;
}

// ---------------- kernel 0b: repack weights [K][256] f32 -> bf16 ------------
// dst[(kb*256 + cout)*32 + q(kk)] = w[(kb*32+kk)*256 + cout], q swaps kk bits 3,4; half h holds K = h*8+j and 16+h*8+j.
__global__ __launch_bounds__(256) void k_repack(const float* __restrict__ src,
                                                __bf16* __restrict__ dst, int K) {
    const int i = blockIdx.x * 256 + threadIdx.x;
    if (i >= K * 256) return;
    const int k = i >> 8, co = i & 255;
    dst[((size_t)((k >> 5) * 256 + co)) * 32 + ((k & 7) | ((k & 8) << 1) | ((k & 16) >> 1))] = (__bf16)src[i];
}

// ---------------- kernel 1: yat conv (3x3, 128 -> 256), SAME ----------------
// block = (n, h): M = 64 pixels (full row), N = 256 couts. 8 waves: wave = mi(4) x nh(2),
// each wave: 16 rows x 128 couts = 8 WMMA tiles. Output stored as bf16 (consumer quantizes
// to bf16 anyway) to halve the dominant HBM stream.
__global__ __launch_bounds__(256) void k_yat(const float* __restrict__ x,
                                             const __bf16* __restrict__ wb,
                                             const float* __restrict__ wsq,
                                             const float* __restrict__ alpha_p,
                                             __bf16* __restrict__ y) {
    extern __shared__ char smem_raw[];
    __bf16* xs = (__bf16*)smem_raw;                      // [3][66][136] bf16 (halo cols, padded)
    float*  s  = (float*)(smem_raw + 3 * 66 * 136 * 2);  // [3][66] per-pixel sum of x^2

    const int tid = threadIdx.x;
    const int n = blockIdx.x >> 6;
    const int h = blockIdx.x & 63;

    for (int i = tid; i < 3 * 66; i += 256) s[i] = 0.f;
    __syncthreads();

    // stage rows h-1..h+1, cols -1..64 (zero pad), 128 ch; accumulate channel sum-of-squares
    for (int i = tid; i < 3 * 66 * 32; i += 256) {
        const int rc = i >> 5;
        const int c4 = (i & 31) << 2;
        const int r  = rc / 66;
        const int wc = rc - r * 66;
        const int gr = h + r - 1;
        const int gc = wc - 1;
        float4 v = make_float4(0.f, 0.f, 0.f, 0.f);
        if ((unsigned)gr < 64u && (unsigned)gc < 64u)
            v = *(const float4*)(x + (((size_t)n * 64 + gr) * 64 + gc) * 128 + c4);
        __bf16* d = xs + (r * 66 + wc) * 136 + c4;
        d[0] = (__bf16)v.x; d[1] = (__bf16)v.y; d[2] = (__bf16)v.z; d[3] = (__bf16)v.w;
        atomicAdd(&s[r * 66 + wc], v.x * v.x + v.y * v.y + v.z * v.z + v.w * v.w);
    }
    __syncthreads();

    const int lane = tid & 31;
    const int wave = tid >> 5;
    const int mi   = wave & 3;
    const int nh   = wave >> 2;
    const int half = lane >> 4;
    const int ml   = lane & 15;
    const int m    = mi * 16 + ml;  // pixel (A row)

    v8f acc[8] = {};

#pragma unroll
    for (int kh = 0; kh < 3; ++kh) {
#pragma unroll
        for (int kw = 0; kw < 3; ++kw) {
            const int tap = kh * 3 + kw;
            const __bf16* arow = xs + (kh * 66 + m + kw) * 136;
            const __bf16* wtap = wb + (size_t)(tap * 4) * 256 * 32;
            __builtin_prefetch(wtap + (size_t)4 * 256 * 32, 0, 0);
            for (int c0 = 0; c0 < 128; c0 += 32) {
                const v16bf a = make_afrag(arow + c0 + half * 8);
                const __bf16* bp =
                    wtap + ((size_t)(c0 >> 5) * 256 + nh * 128 + ml) * 32 + half * 16;
#pragma unroll
                for (int t = 0; t < 8; ++t) {
                    const v16bf b = *(const v16bf*)(bp + (size_t)t * 16 * 32);
                    acc[t] = WMMA_BF16(a, b, acc[t]);
                }
            }
        }
    }

    // epilogue: y = dot^2 / (||patch||^2 + ||w||^2 - 2 dot + eps) * scale
    const float scale = powf(16.0f / log1pf(256.0f), alpha_p[0]);
    float ps[8];
#pragma unroll
    for (int i = 0; i < 8; ++i) {
        const int mr = mi * 16 + half * 8 + i;  // C/D row = pixel
        float v = 0.f;
#pragma unroll
        for (int kh = 0; kh < 3; ++kh)
#pragma unroll
            for (int kw = 0; kw < 3; ++kw) v += s[kh * 66 + mr + kw];
        ps[i] = v;
    }
    __bf16* orow = y + (((size_t)n * 64 + h) * 64) * 256;
#pragma unroll
    for (int t = 0; t < 8; ++t) {
        const int co = nh * 128 + t * 16 + ml;
        const float wq = wsq[co];
#pragma unroll
        for (int i = 0; i < 8; ++i) {
            const int mr = mi * 16 + half * 8 + i;
            const float dot  = acc[t][i];
            const float dist = ps[i] + wq - 2.0f * dot + 1e-5f;
            orow[(size_t)mr * 256 + co] = (__bf16)(dot * dot / dist * scale);
        }
    }
}

// ------- kernel 2: conv3x3(y,w_lin) + 1x1(x,w_proj) + avgpool 2x2 -----------
// block = (n, oh, ow-half): pre-pool tile 2 rows x 32 cols (M=64), N=256.
// y halo (bf16, pure byte copy) is staged via the Tensor Data Mover: 4 per-row
// descriptors with hardware LDS padding; OOB border chunks zero-filled by VALU.
__global__ __launch_bounds__(256) void k_conv_pool(const float* __restrict__ x,
                                                   const __bf16* __restrict__ y,
                                                   const __bf16* __restrict__ wlb,
                                                   const __bf16* __restrict__ wpb,
                                                   float* __restrict__ out) {
    extern __shared__ char smem_raw[];
    __bf16* ys  = (__bf16*)smem_raw;                       // [4][34][264] bf16 y halo (offset 0!)
    __bf16* xs2 = (__bf16*)(smem_raw + 4 * 34 * 264 * 2);  // [64][136] bf16 x tile
    float*  y2  = (float*)smem_raw;                        // reuse after GEMM: [64][260] f32

    const int tid = threadIdx.x;
    const int n   = blockIdx.x >> 6;
    const int oh  = (blockIdx.x >> 1) & 31;
    const int owh = blockIdx.x & 1;
    const int r0  = oh * 2;    // pre-pool row base
    const int cb  = owh * 32;  // pre-pool col base

    // zero-fill OOB border chunks of the y halo (disjoint from TDM-written region)
    for (int i = tid; i < 4 * 34 * 32; i += 256) {  // 16B units: (r,cc) x 32 x uint4
        const int rc = i >> 5;
        const int r  = rc / 34;
        const int cc = rc - r * 34;
        const int gy = r0 - 1 + r;
        const int gx = cb - 1 + cc;
        if ((unsigned)gy >= 64u || (unsigned)gx >= 64u) {
            uint4* d = (uint4*)((char*)(ys + (r * 34 + cc) * 264) + (i & 31) * 16);
            *d = make_uint4(0u, 0u, 0u, 0u);
        }
    }
    // stage x rows r0..r0+1, cols cb..cb+31, 128 ch (needs f32->bf16, stays on VALU)
    for (int i = tid; i < 64 * 32; i += 256) {
        const int p  = i >> 5;
        const int c4 = (i & 31) << 2;
        float4 v = *(const float4*)(
            x + (((size_t)n * 64 + r0 + (p >> 5)) * 64 + cb + (p & 31)) * 128 + c4);
        __bf16* d = xs2 + p * 136 + c4;
        d[0] = (__bf16)v.x; d[1] = (__bf16)v.y; d[2] = (__bf16)v.z; d[3] = (__bf16)v.w;
    }

    // wave 0 issues the TDM descriptors for the in-range y halo rows
    if (tid < 32) {
        const int cc_start = owh ? 0 : 1;   // which halo column is OOB
        const int gx0 = cb - 1 + cc_start;  // first in-range global column (33 in-range cols)
#pragma unroll
        for (int r = 0; r < 4; ++r) {
            const int gy = r0 - 1 + r;
            if ((unsigned)gy < 64u) {
                const __bf16* gsrc = y + (((size_t)n * 64 + gy) * 64 + gx0) * 256;
                tdm_load_row((unsigned)((r * 34 + cc_start) * 264 * 2), gsrc, 33 * 256);
            }
        }
        __builtin_amdgcn_s_wait_tensorcnt(0);
    }
    __syncthreads();

    const int lane = tid & 31;
    const int wave = tid >> 5;
    const int mi = wave & 3, nh = wave >> 2;
    const int half = lane >> 4, ml = lane & 15;
    const int m  = mi * 16 + ml;
    const int pr = m >> 5, pc = m & 31;

    v8f acc[8] = {};

#pragma unroll
    for (int kh = 0; kh < 3; ++kh) {
#pragma unroll
        for (int kw = 0; kw < 3; ++kw) {
            const int tap = kh * 3 + kw;
            const __bf16* arow = ys + ((pr + kh) * 34 + pc + kw) * 264;
            const __bf16* wtap = wlb + (size_t)(tap * 8) * 256 * 32;
            __builtin_prefetch(wtap + (size_t)8 * 256 * 32, 0, 0);
            for (int c0 = 0; c0 < 256; c0 += 32) {
                const v16bf a = make_afrag(arow + c0 + half * 8);
                const __bf16* bp =
                    wtap + ((size_t)(c0 >> 5) * 256 + nh * 128 + ml) * 32 + half * 16;
#pragma unroll
                for (int t = 0; t < 8; ++t) {
                    const v16bf b = *(const v16bf*)(bp + (size_t)t * 16 * 32);
                    acc[t] = WMMA_BF16(a, b, acc[t]);
                }
            }
        }
    }
    {  // residual 1x1 projection, K = 128
        const __bf16* arow = xs2 + (pr * 32 + pc) * 136;
        for (int c0 = 0; c0 < 128; c0 += 32) {
            const v16bf a = make_afrag(arow + c0 + half * 8);
            const __bf16* bp = wpb + ((size_t)(c0 >> 5) * 256 + nh * 128 + ml) * 32 + half * 16;
#pragma unroll
            for (int t = 0; t < 8; ++t) {
                const v16bf b = *(const v16bf*)(bp + (size_t)t * 16 * 32);
                acc[t] = WMMA_BF16(a, b, acc[t]);
            }
        }
    }
    __syncthreads();  // all waves done reading ys before region is reused

    // spill tile to LDS for cross-wave 2x2 pooling
#pragma unroll
    for (int t = 0; t < 8; ++t) {
        const int co = nh * 128 + t * 16 + ml;
#pragma unroll
        for (int i = 0; i < 8; ++i) {
            const int mr = mi * 16 + half * 8 + i;
            y2[mr * 260 + co] = acc[t][i];
        }
    }
    __syncthreads();

    float* ob = out + (((size_t)n * 32 + oh) * 32 + owh * 16) * 256;
    for (int i = tid; i < 16 * 256; i += 256) {
        const int p  = i >> 8;
        const int ch = i & 255;
        const float v = 0.25f * (y2[(2 * p) * 260 + ch] + y2[(2 * p + 1) * 260 + ch] +
                                 y2[(2 * p + 32) * 260 + ch] + y2[(2 * p + 33) * 260 + ch]);
        ob[(size_t)p * 256 + ch] = v;
    }
}

extern "C" void kernel_launch(void* const* d_in, const int* in_sizes, int n_in,
                              void* d_out, int out_size, void* d_ws, size_t ws_size,
                              hipStream_t stream) {
    const float* x      = (const float*)d_in[0];
    const float* w_yat  = (const float*)d_in[1];
    const float* alpha  = (const float*)d_in[2];
    const float* w_lin  = (const float*)d_in[3];
    const float* w_proj = (const float*)d_in[4];
    float* out = (float*)d_out;

    char* ws = (char*)d_ws;
    float*  wsq = (float*)ws;                                   // 256 f32
    __bf16* wyb = (__bf16*)(ws + 4096);                         // 1152*256 bf16
    __bf16* wlb = (__bf16*)(ws + 4096 + 589824);                // 2304*256 bf16
    __bf16* wpb = (__bf16*)(ws + 4096 + 589824 + 1179648);      //  128*256 bf16
    __bf16* ybuf = (__bf16*)(ws + 2097152);                     // 32*64*64*256 bf16

    k_wsq<<<1, 256, 0, stream>>>(w_yat, wsq);
    k_repack<<<(1152 * 256) / 256, 256, 0, stream>>>(w_yat, wyb, 1152);
    k_repack<<<(2304 * 256) / 256, 256, 0, stream>>>(w_lin, wlb, 2304);
    k_repack<<<(128 * 256) / 256, 256, 0, stream>>>(w_proj, wpb, 128);

    const size_t lds1 = (size_t)3 * 66 * 136 * 2 + (size_t)3 * 66 * 4;
    k_yat<<<32 * 64, 256, lds1, stream>>>(x, wyb, wsq, alpha, ybuf);

    const size_t lds2 = (size_t)4 * 34 * 264 * 2 + (size_t)64 * 136 * 2;
    k_conv_pool<<<32 * 32 * 2, 256, lds2, stream>>>(x, ybuf, wlb, wpb, out);
}
